// BertSelfAttention_90417651516422
// MI455X (gfx1250) — matrix operand order repaired and checked by reading, not compiled, not run on hardware
//
#include <hip/hip_runtime.h>
#include <math.h>
#include <stdint.h>

// ---------------------------------------------------------------------------
// BERT self-attention forward for MI455X (gfx1250, wave32, WMMA + TDM).
// B=4, S=2048, D=768, H=12, DH=64. f16 WMMA, f32 accumulate.
// ---------------------------------------------------------------------------

#define BB   4
#define SS   2048
#define DD   768
#define HH   12
#define DHD  64

typedef __attribute__((ext_vector_type(16))) _Float16 v16h;
typedef __attribute__((ext_vector_type(8)))  _Float16 v8h;
typedef __attribute__((ext_vector_type(8)))  float    v8f;
typedef __attribute__((ext_vector_type(4)))  unsigned int u32x4;
typedef __attribute__((ext_vector_type(8)))  int          i32x8;
typedef __attribute__((ext_vector_type(4)))  int          i32x4;

static __device__ __forceinline__ v16h hcat(v8h lo, v8h hi) {
  return __builtin_shufflevector(lo, hi, 0,1,2,3,4,5,6,7,8,9,10,11,12,13,14,15);
}

static __device__ __forceinline__ v8f wmma_f16(v16h a, v16h b, v8f c) {
  // (neg_a, A, neg_b, B, c_mod, C, reuse_a, reuse_b)
  return __builtin_amdgcn_wmma_f32_16x16x32_f16(false, a, false, b, (short)0, c,
                                                false, false);
}

#define HAVE_TDM __has_builtin(__builtin_amdgcn_tensor_load_to_lds)

#if HAVE_TDM
// Issue a TDM 2D tile load: global (f16, row stride 'stride0' elems) -> LDS,
// written contiguously (tile_d0 elems per row, tile_d1 rows).
// D# packing per CDNA5 ISA 8.3/8.4 (count=1, type=2, data_size=2B, 2D).
static __device__ __forceinline__ void tdm_load_2d(
    unsigned lds_addr, const _Float16* gaddr,
    unsigned tensor_d0, unsigned tensor_d1,
    unsigned tile_d0, unsigned tile_d1, unsigned stride0)
{
  unsigned long long ga = (unsigned long long)(uintptr_t)gaddr;
  u32x4 g0;
  g0.x = 1u;                                          // count=1, user mode
  g0.y = lds_addr;                                    // lds_addr [63:32]
  g0.z = (unsigned)(ga & 0xFFFFFFFFu);                // global_addr lo
  g0.w = (unsigned)((ga >> 32) & 0x01FFFFFFu) | (2u << 30);  // addr hi | type=2
  i32x8 g1;
  g1[0] = (int)(1u << 16);                            // data_size=1 (2 bytes)
  g1[1] = (int)((tensor_d0 & 0xFFFFu) << 16);         // tensor_dim0 [79:48]
  g1[2] = (int)((tensor_d0 >> 16) | ((tensor_d1 & 0xFFFFu) << 16));
  g1[3] = (int)((tensor_d1 >> 16) | (tile_d0 << 16)); // tile_dim0 [127:112]
  g1[4] = (int)(tile_d1 & 0xFFFFu);                   // tile_dim1; tile_dim2=0
  g1[5] = (int)stride0;                               // tensor_dim0_stride lo
  g1[6] = 0;                                          // stride0 hi | stride1 lo
  g1[7] = 0;                                          // stride1 hi
  i32x4 gz = {0, 0, 0, 0};
#if defined(__clang_major__) && __clang_major__ >= 23
  i32x8 gz8 = {0, 0, 0, 0, 0, 0, 0, 0};
  __builtin_amdgcn_tensor_load_to_lds(g0, g1, gz, gz, gz8, 0);
#else
  __builtin_amdgcn_tensor_load_to_lds(g0, g1, gz, gz, 0);
#endif
}
#endif

// ---------------------------------------------------------------------------
// Kernel 1: fused QKV projection.
//   out = (X @ W + b) [* 0.125 for Q].
//   Q,K stored f16 split-head [B,H,S,64]; V stored transposed [B,H,64,S]
//   so the attention kernel can stage V^T tiles with plain 2D TDM loads.
// ---------------------------------------------------------------------------
__global__ __launch_bounds__(256) void qkv_proj_kernel(
    const float* __restrict__ X,
    const float* __restrict__ Wq, const float* __restrict__ bq,
    const float* __restrict__ Wk, const float* __restrict__ bk,
    const float* __restrict__ Wv, const float* __restrict__ bv,
    _Float16* __restrict__ q_ws, _Float16* __restrict__ k_ws,
    _Float16* __restrict__ v_ws)
{
  const int z = blockIdx.z;
  const float* W    = (z == 0) ? Wq : (z == 1) ? Wk : Wv;
  const float* bias = (z == 0) ? bq : (z == 1) ? bk : bv;
  _Float16*    ows  = (z == 0) ? q_ws : (z == 1) ? k_ws : v_ws;
  const float  oscale = (z == 0) ? 0.125f : 1.0f;   // 1/sqrt(DH) folded into Q

  const int rowbase = blockIdx.x * 64;
  const int hd      = blockIdx.y;          // head index; nbase = hd*64
  const int nbase   = hd * 64;

  const int t    = threadIdx.x;
  const int wave = t >> 5;
  const int lane = t & 31;
  const int hf   = lane >> 4;              // lane half (K striping)
  const int ln   = lane & 15;
  const int wr   = wave >> 1;              // 0..3 row sub-tile
  const int wc   = wave & 1;               // 0..1 col sub-tile (32 cols)

  __shared__ __align__(16) _Float16 Xt[64][32];   // [row][k]
  __shared__ __align__(16) _Float16 Wt[64][32];   // [n][k] (transposed W tile)

  const v8f vzero = {0.f,0.f,0.f,0.f,0.f,0.f,0.f,0.f};
  v8f acc0 = vzero, acc1 = vzero;

  for (int kb = 0; kb < DD; kb += 32) {
    __syncthreads();
#pragma unroll
    for (int i = 0; i < 8; ++i) {
      int idx = t + i * 256;
      int r = idx >> 5, c = idx & 31;
      Xt[r][c] = (_Float16)X[(size_t)(rowbase + r) * DD + kb + c];
    }
#pragma unroll
    for (int i = 0; i < 8; ++i) {
      int idx = t + i * 256;
      int k = idx >> 6, n = idx & 63;
      Wt[n][k] = (_Float16)W[(size_t)(kb + k) * DD + nbase + n];
    }
    __syncthreads();

    const int arow = wr * 16 + ln;
    v16h a = hcat(*(const v8h*)&Xt[arow][hf * 16],
                  *(const v8h*)&Xt[arow][hf * 16 + 8]);
    const int b0n = wc * 32 + ln;
    v16h b0 = hcat(*(const v8h*)&Wt[b0n][hf * 16],
                   *(const v8h*)&Wt[b0n][hf * 16 + 8]);
    v16h b1 = hcat(*(const v8h*)&Wt[b0n + 16][hf * 16],
                   *(const v8h*)&Wt[b0n + 16][hf * 16 + 8]);
    acc0 = wmma_f16(a, b0, acc0);
    acc1 = wmma_f16(a, b1, acc1);
  }

  const int dh0 = wc * 32 + ln;         // 0..47
  const int dh1 = dh0 + 16;             // 16..63
  const float bia0 = bias[nbase + dh0];
  const float bia1 = bias[nbase + dh1];
#pragma unroll
  for (int r = 0; r < 8; ++r) {
    int rowg = rowbase + wr * 16 + r + 8 * hf;
    int b = rowg >> 11;                 // /2048
    int s = rowg & 2047;
    _Float16 o0 = (_Float16)((acc0[r] + bia0) * oscale);
    _Float16 o1 = (_Float16)((acc1[r] + bia1) * oscale);
    if (z == 2) {
      // V transposed: [B,H,DH,S]
      size_t hb = ((size_t)b * HH + hd) * DHD;
      ows[(hb + dh0) * SS + s] = o0;
      ows[(hb + dh1) * SS + s] = o1;
    } else {
      size_t base = (((size_t)b * HH + hd) * SS + s) * DHD;
      ows[base + dh0] = o0;
      ows[base + dh1] = o1;
    }
  }
}

// ---------------------------------------------------------------------------
// Kernel 2: causal flash attention per (b,h).
// 128 threads = 4 waves x 16 query rows. Keys in blocks of 32; K and V^T
// tiles staged into double-buffered LDS by the Tensor Data Mover (wave 0
// issues, TENSORcnt-synced), overlapping DMA with WMMA of the current block.
// ---------------------------------------------------------------------------
__global__ __launch_bounds__(128) void attn_kernel(
    const _Float16* __restrict__ q_ws,
    const _Float16* __restrict__ k_ws,
    const _Float16* __restrict__ v_ws,
    float* __restrict__ out)
{
  const int bh    = blockIdx.y;         // 0..47
  const int b     = bh / HH;
  const int hd    = bh % HH;
  const int qbase = blockIdx.x * 64;

  const int t    = threadIdx.x;
  const int wv   = t >> 5;              // wave 0..3
  const int lane = t & 31;
  const int hf   = lane >> 4;
  const int ln   = lane & 15;

  __shared__ __align__(16) _Float16 KtB[2][32][64];   // [buf][key][dh]
  __shared__ __align__(16) _Float16 VtB[2][64][32];   // [buf][dh][key]
  __shared__ __align__(16) _Float16 Pt[4][16][32];    // per-wave P [q][key]

  // Q fragments held in registers for the whole kernel.
  const _Float16* qp = q_ws + ((size_t)bh * SS) * DHD;
  const int qrow = qbase + wv * 16 + ln;              // A-layout row = lane%16
  const _Float16* qr = qp + (size_t)qrow * DHD;
  v16h qa0 = hcat(*(const v8h*)(qr + hf * 16),        // dh 0..31
                  *(const v8h*)(qr + hf * 16 + 8));
  v16h qa1 = hcat(*(const v8h*)(qr + 32 + hf * 16),   // dh 32..63
                  *(const v8h*)(qr + 32 + hf * 16 + 8));

  const v8f vzero = {0.f,0.f,0.f,0.f,0.f,0.f,0.f,0.f};
  v8f acc[4] = {vzero, vzero, vzero, vzero};          // ctx [16q x 64dh]
  float m_r[8], l_r[8];
#pragma unroll
  for (int r = 0; r < 8; ++r) { m_r[r] = -1.0e30f; l_r[r] = 0.0f; }

  const int qrow0   = qbase + wv * 16 + 8 * hf;       // C-layout row base
  const int nblocks = qbase / 32 + 2;                 // keys 0 .. qbase+63

  const _Float16* kbh = k_ws + (size_t)bh * SS * DHD;  // [S][64]
  const _Float16* vbh = v_ws + (size_t)bh * DHD * SS;  // [64][S] (transposed)

#if HAVE_TDM
  // Stage key-block jb into LDS buffer 'buf' via the Tensor Data Mover.
  auto stage = [&](int jb, int buf) {
    tdm_load_2d((unsigned)(uintptr_t)&KtB[buf][0][0], kbh + (size_t)jb * DHD,
                /*tensor_d0=*/DHD, /*tensor_d1=*/SS,
                /*tile_d0=*/DHD, /*tile_d1=*/32, /*stride0=*/DHD);
    tdm_load_2d((unsigned)(uintptr_t)&VtB[buf][0][0], vbh + jb,
                /*tensor_d0=*/SS, /*tensor_d1=*/DHD,
                /*tile_d0=*/32, /*tile_d1=*/DHD, /*stride0=*/SS);
  };
  if (wv == 0) stage(0, 0);            // prologue: prefetch block 0
#endif

  for (int jb_i = 0; jb_i < nblocks; ++jb_i) {
    const int jb  = jb_i * 32;
    const int buf = jb_i & 1;

#if HAVE_TDM
    if (wv == 0) __builtin_amdgcn_s_wait_tensorcnt((short)0);  // block jb landed
    __syncthreads();                    // all waves: tile visible; prev buf free
    if (wv == 0 && (jb_i + 1) < nblocks) stage(jb + 32, buf ^ 1);
#else
    __syncthreads();
    // Fallback: manual cooperative staging (16-byte chunks).
#pragma unroll
    for (int i = 0; i < 2; ++i) {
      int c = t + i * 128;
      int key = c >> 3, dg = c & 7;
      *(v8h*)&KtB[buf][key][dg * 8] =
          *(const v8h*)(kbh + (size_t)(jb + key) * DHD + dg * 8);
    }
#pragma unroll
    for (int i = 0; i < 2; ++i) {
      int c = t + i * 128;
      int dh = c >> 2, kg = (c & 3) * 8;
      *(v8h*)&VtB[buf][dh][kg] =
          *(const v8h*)(vbh + (size_t)dh * SS + jb + kg);
    }
    __syncthreads();
#endif

    // Scores: S = Q @ K^T (two 16-key tiles, contraction dh in two k-steps).
    v8f s0 = vzero, s1 = vzero;
    {
      v16h bk = hcat(*(const v8h*)&KtB[buf][ln][hf * 16],
                     *(const v8h*)&KtB[buf][ln][hf * 16 + 8]);
      s0 = wmma_f16(qa0, bk, s0);
      bk = hcat(*(const v8h*)&KtB[buf][ln][32 + hf * 16],
                *(const v8h*)&KtB[buf][ln][32 + hf * 16 + 8]);
      s0 = wmma_f16(qa1, bk, s0);
      bk = hcat(*(const v8h*)&KtB[buf][16 + ln][hf * 16],
                *(const v8h*)&KtB[buf][16 + ln][hf * 16 + 8]);
      s1 = wmma_f16(qa0, bk, s1);
      bk = hcat(*(const v8h*)&KtB[buf][16 + ln][32 + hf * 16],
                *(const v8h*)&KtB[buf][16 + ln][32 + hf * 16 + 8]);
      s1 = wmma_f16(qa1, bk, s1);
    }

    // Online softmax (rows split across lane halves, matching C-layout).
#pragma unroll
    for (int r = 0; r < 8; ++r) {
      float v0 = s0[r], v1 = s1[r];
      const int q  = qrow0 + r;
      const int k0 = jb + ln;
      v0 = (k0      <= q) ? v0 : -1.0e30f;   // causal mask
      v1 = (k0 + 16 <= q) ? v1 : -1.0e30f;
      float mx = fmaxf(v0, v1);
      mx = fmaxf(mx, __shfl_xor(mx, 1));
      mx = fmaxf(mx, __shfl_xor(mx, 2));
      mx = fmaxf(mx, __shfl_xor(mx, 4));
      mx = fmaxf(mx, __shfl_xor(mx, 8));
      const float mn = fmaxf(m_r[r], mx);
      const float al = __expf(m_r[r] - mn);
      const float p0 = __expf(v0 - mn);
      const float p1 = __expf(v1 - mn);
      float sm = p0 + p1;
      sm += __shfl_xor(sm, 1);
      sm += __shfl_xor(sm, 2);
      sm += __shfl_xor(sm, 4);
      sm += __shfl_xor(sm, 8);
      l_r[r] = l_r[r] * al + sm;
      m_r[r] = mn;
      acc[0][r] *= al; acc[1][r] *= al; acc[2][r] *= al; acc[3][r] *= al;
      Pt[wv][r + 8 * hf][ln]      = (_Float16)p0;
      Pt[wv][r + 8 * hf][16 + ln] = (_Float16)p1;
    }
    __syncthreads();   // cross-lane P visibility (uniform control flow)

    // ctx += P @ V  (A = P 16x32 from LDS, B from transposed V tile).
    v16h pa = hcat(*(const v8h*)&Pt[wv][ln][hf * 16],
                   *(const v8h*)&Pt[wv][ln][hf * 16 + 8]);
#pragma unroll
    for (int dt = 0; dt < 4; ++dt) {
      v16h bv_ = hcat(*(const v8h*)&VtB[buf][dt * 16 + ln][hf * 16],
                      *(const v8h*)&VtB[buf][dt * 16 + ln][hf * 16 + 8]);
      acc[dt] = wmma_f16(pa, bv_, acc[dt]);
    }
  }

  // Epilogue: normalize and store f32 [B,S,H*DH].
#pragma unroll
  for (int r = 0; r < 8; ++r) {
    const float inv = 1.0f / l_r[r];
    const int q = qrow0 + r;
    const size_t ob = ((size_t)b * SS + q) * DD + hd * DHD;
#pragma unroll
    for (int dt = 0; dt < 4; ++dt)
      out[ob + dt * 16 + ln] = acc[dt][r] * inv;
  }
}

// ---------------------------------------------------------------------------
extern "C" void kernel_launch(void* const* d_in, const int* in_sizes, int n_in,
                              void* d_out, int out_size, void* d_ws, size_t ws_size,
                              hipStream_t stream) {
  (void)in_sizes; (void)n_in; (void)out_size; (void)ws_size;
  const float* X  = (const float*)d_in[0];
  // d_in[1] = attention_mask (zeros, unused by the reference math)
  const float* Wq = (const float*)d_in[2];
  const float* bq = (const float*)d_in[3];
  const float* Wk = (const float*)d_in[4];
  const float* bk = (const float*)d_in[5];
  const float* Wv = (const float*)d_in[6];
  const float* bv = (const float*)d_in[7];
  float* out = (float*)d_out;

  // Workspace: 3 x [B,H,S,64] f16 = 37.75 MB (V stored transposed).
  const size_t per = (size_t)BB * HH * SS * DHD;     // 6,291,456 elems
  _Float16* q_ws = (_Float16*)d_ws;
  _Float16* k_ws = q_ws + per;
  _Float16* v_ws = k_ws + per;

  dim3 g1((BB * SS) / 64, DD / 64, 3);   // 128 x 12 x 3
  qkv_proj_kernel<<<g1, dim3(256), 0, stream>>>(X, Wq, bq, Wk, bk, Wv, bv,
                                                q_ws, k_ws, v_ws);

  dim3 g2(SS / 64, BB * HH);             // 32 x 48
  attn_kernel<<<g2, dim3(128), 0, stream>>>(q_ws, k_ws, v_ws, out);
}
